// Attention_35588099015470
// MI455X (gfx1250) — matrix operand order repaired and checked by reading, compiled, not run on hardware
//
#include <hip/hip_runtime.h>
#include <hip/hip_bf16.h>
#include <math.h>

// ---------------------------------------------------------------------------
// Problem constants (B, T, C, H, R) = (2, 2048, 2048, 16, 8), D = 128
// ---------------------------------------------------------------------------
#define B_SZ 2
#define T_SZ 2048
#define C_SZ 2048
#define H_SZ 16
#define D_SZ 128
#define R_SZ 8
#define M_ROWS (B_SZ * T_SZ) // 4096

typedef __attribute__((ext_vector_type(16))) __bf16 v16bf;
typedef __attribute__((ext_vector_type(8)))  __bf16 v8bf;
typedef __attribute__((ext_vector_type(8)))  float  v8f;

union bfu { v16bf v; v8bf h[2]; };

// ---------------------------------------------------------------------------
// fp32 -> bf16 convert
// ---------------------------------------------------------------------------
__global__ void cvt_f32_bf16(const float* __restrict__ src, __bf16* __restrict__ dst) {
    const int i = blockIdx.x * blockDim.x + threadIdx.x;
    dst[i] = (__bf16)src[i];
}

// ---------------------------------------------------------------------------
// Wt[n][k] = w[n][k] + use_lora * sum_r A[k][r] * Bm[r][n]   (bf16, transposed
// so the GEMM reads contiguous K for B-fragments)
// ---------------------------------------------------------------------------
__global__ void prep_w(const float* __restrict__ w, const float* __restrict__ A,
                       const float* __restrict__ Bm, const int* __restrict__ use_lora,
                       __bf16* __restrict__ Wt) {
    const int idx = blockIdx.x * blockDim.x + threadIdx.x;    // 0 .. 2048*2048-1
    const int k = idx & (C_SZ - 1);
    const int n = idx >> 11;
    float v = w[(size_t)n * C_SZ + k];
    if (use_lora[0]) {
        float s = 0.f;
#pragma unroll
        for (int r = 0; r < R_SZ; ++r)
            s += A[(size_t)k * R_SZ + r] * Bm[(size_t)r * C_SZ + n];
        v += s;
    }
    Wt[(size_t)n * C_SZ + k] = (__bf16)v;
}

// ---------------------------------------------------------------------------
// WMMA bf16 GEMM: Y[m][n] = sum_k X[m][k]*Wt[n][k] + bias[n]
// block tile 128x128, 8 waves (2x4), wave tile 64x32, BK = 32
// ---------------------------------------------------------------------------
template <typename OutT>
__global__ __launch_bounds__(256, 1)
void gemm_bf16(const __bf16* __restrict__ X, const __bf16* __restrict__ Wt,
               const float* __restrict__ bias, OutT* __restrict__ Y,
               int M, int N, int K) {
    __shared__ __align__(16) __bf16 As[128][40]; // [m][k], padded
    __shared__ __align__(16) __bf16 Bs[128][40]; // [n][k], padded

    const int tid = threadIdx.x;
    const int w  = tid >> 5;
    const int l  = tid & 31;
    const int ln = l & 15;
    const int hi = l >> 4;            // 0: lanes 0-15, 1: lanes 16-31
    const int wm = w >> 2;            // 0..1
    const int wn = w & 3;             // 0..3
    const int bm = blockIdx.y * 128;
    const int bn = blockIdx.x * 128;

    const int kbA = hi ? 8 : 0;       // A-fragment K chunk base
    const int kbB = hi ? 8 : 0;       // B-fragment K base

    v8f acc[4][2] = {};

    for (int kt = 0; kt < K; kt += 32) {
        __syncthreads();
        {   // cooperative tile loads: 16 bf16 per thread per tile
            const int r  = tid >> 1;
            const int c0 = (tid & 1) * 16;
            const __bf16* sa = X  + (size_t)(bm + r) * K + kt + c0;
            *(v8bf*)&As[r][c0]     = *(const v8bf*)sa;
            *(v8bf*)&As[r][c0 + 8] = *(const v8bf*)(sa + 8);
            const __bf16* sb = Wt + (size_t)(bn + r) * K + kt + c0;
            *(v8bf*)&Bs[r][c0]     = *(const v8bf*)sb;
            *(v8bf*)&Bs[r][c0 + 8] = *(const v8bf*)(sb + 8);
        }
        __syncthreads();

        v16bf af[4];
#pragma unroll
        for (int ms = 0; ms < 4; ++ms) {
            const int r = wm * 64 + ms * 16 + ln;
            bfu u;
            u.h[0] = *(const v8bf*)&As[r][kbA];
            u.h[1] = *(const v8bf*)&As[r][kbA + 16];
            af[ms] = u.v;
        }
#pragma unroll
        for (int ns = 0; ns < 2; ++ns) {
            const int r = wn * 32 + ns * 16 + ln;
            bfu u;
            u.h[0] = *(const v8bf*)&Bs[r][kbB];
            u.h[1] = *(const v8bf*)&Bs[r][kbB + 16];
            const v16bf bf = u.v;
#pragma unroll
            for (int ms = 0; ms < 4; ++ms)
                acc[ms][ns] = __builtin_amdgcn_wmma_f32_16x16x32_bf16(
                    false, af[ms], false, bf, (short)0, acc[ms][ns], false, false);
        }
    }

#pragma unroll
    for (int ns = 0; ns < 2; ++ns) {
        const int n = bn + wn * 32 + ns * 16 + ln;
        const float bv = bias[n];
#pragma unroll
        for (int ms = 0; ms < 4; ++ms)
#pragma unroll
            for (int j = 0; j < 8; ++j) {
                const int m = bm + wm * 64 + ms * 16 + j + hi * 8;
                const float v = acc[ms][ns][j] + bv;
                Y[(size_t)m * N + n] = (OutT)v;
            }
    }
}

// ---------------------------------------------------------------------------
// RoPE: read [b*T+t][h*D+d] bf16, write head-major [bh][t][d] bf16.
// scale = 1/sqrt(D) folded into Q (1.0 for K).
// ---------------------------------------------------------------------------
__global__ void rope_kernel(const __bf16* __restrict__ src, const float* __restrict__ cosb,
                            const float* __restrict__ sinb, __bf16* __restrict__ dst,
                            float scale) {
    const int idx = blockIdx.x * blockDim.x + threadIdx.x;    // B*T*H*64 threads
    const int i = idx & 63;
    const int h = (idx >> 6) & 15;
    const int t = (idx >> 10) & 2047;
    const int b = idx >> 21;
    const size_t s = ((size_t)(b * T_SZ + t)) * C_SZ + h * D_SZ + 2 * i;
    const float x1 = (float)src[s];
    const float x2 = (float)src[s + 1];
    const float c = cosb[(size_t)t * 64 + i];
    const float sn = sinb[(size_t)t * 64 + i];
    const size_t o = ((size_t)(b * H_SZ + h) * T_SZ + t) * D_SZ + 2 * i;
    dst[o]     = (__bf16)((x1 * c - x2 * sn) * scale);
    dst[o + 1] = (__bf16)((x1 * sn + x2 * c) * scale);
}

// ---------------------------------------------------------------------------
// V transpose: [b*T+t][h*D+d] bf16 -> Vt[bh][d][t] bf16 (for contiguous AV
// B-fragments along the key dimension)
// ---------------------------------------------------------------------------
__global__ void v_transpose(const __bf16* __restrict__ src, __bf16* __restrict__ dst) {
    const int idx = blockIdx.x * blockDim.x + threadIdx.x;    // B*H*D*T threads
    const int t = idx & 2047;
    const int d = (idx >> 11) & 127;
    const int h = (idx >> 18) & 15;
    const int b = idx >> 22;
    dst[((size_t)(b * H_SZ + h) * D_SZ + d) * T_SZ + t] =
        src[((size_t)(b * T_SZ + t)) * C_SZ + h * D_SZ + d];
}

// ---------------------------------------------------------------------------
// Flash attention: grid (T/128, B*H), 256 threads = 8 waves.
// Each wave owns 16 query rows; loops over 32-key tiles with online softmax.
// Q pre-scaled by 1/sqrt(D). Writes Y[b*T+t][h*D+d] bf16.
// ---------------------------------------------------------------------------
__global__ __launch_bounds__(256, 1)
void attn_kernel(const __bf16* __restrict__ Qh, const __bf16* __restrict__ Kh,
                 const __bf16* __restrict__ Vt, __bf16* __restrict__ Yb) {
    __shared__ __align__(16) __bf16 Ks[32][136];   // [key][d]
    __shared__ __align__(16) __bf16 Vts[128][40];  // [d][key-local]
    __shared__ __align__(16) __bf16 Ps[8][16][40]; // per-wave P staging [m][key]

    const int bh = blockIdx.y;
    const int qt = blockIdx.x;
    const int tid = threadIdx.x;
    const int w  = tid >> 5;
    const int l  = tid & 31;
    const int ln = l & 15;
    const int hi = l >> 4;
    const int b = bh >> 4, h = bh & 15;
    const int qrow0 = qt * 128 + w * 16;

    const int kbA = hi ? 8 : 0;   // A-fragment K chunk base
    const int kbB = hi ? 8 : 0;   // B-fragment K base

    // Load Q fragments (row = ln, 4 chunks of K=32 over D=128)
    v16bf qf[4];
    {
        const __bf16* qr = Qh + ((size_t)bh * T_SZ + qrow0 + ln) * D_SZ;
#pragma unroll
        for (int c = 0; c < 4; ++c) {
            bfu u;
            u.h[0] = *(const v8bf*)(qr + c * 32 + kbA);
            u.h[1] = *(const v8bf*)(qr + c * 32 + kbA + 16);
            qf[c] = u.v;
        }
    }

    v8f acc[8] = {};
    float mrun[8], lrun[8];
#pragma unroll
    for (int j = 0; j < 8; ++j) { mrun[j] = -3.0e38f; lrun[j] = 0.f; }

    const int kend = qt * 128 + 128;  // causal bound (uniform per block)
    for (int kb0 = 0; kb0 < kend; kb0 += 32) {
        __syncthreads();
        {   // K tile: 32 keys x 128 d
            const int r  = tid >> 3;
            const int c0 = (tid & 7) * 16;
            const __bf16* s = Kh + ((size_t)bh * T_SZ + kb0 + r) * D_SZ + c0;
            *(v8bf*)&Ks[r][c0]     = *(const v8bf*)s;
            *(v8bf*)&Ks[r][c0 + 8] = *(const v8bf*)(s + 8);
        }
        {   // Vt tile: 128 d x 32 keys
            const int d  = tid >> 1;
            const int c0 = (tid & 1) * 16;
            const __bf16* s = Vt + ((size_t)bh * D_SZ + d) * T_SZ + kb0 + c0;
            *(v8bf*)&Vts[d][c0]     = *(const v8bf*)s;
            *(v8bf*)&Vts[d][c0 + 8] = *(const v8bf*)(s + 8);
        }
        __syncthreads();

        // S = Q K^T for 16 q-rows x 32 keys (2 subtiles x 4 K-chunks)
        v8f S0 = {}, S1 = {};
#pragma unroll
        for (int c = 0; c < 4; ++c) {
            bfu u0, u1;
            const int d0 = c * 32 + kbB;
            u0.h[0] = *(const v8bf*)&Ks[ln][d0];
            u0.h[1] = *(const v8bf*)&Ks[ln][d0 + 16];
            u1.h[0] = *(const v8bf*)&Ks[16 + ln][d0];
            u1.h[1] = *(const v8bf*)&Ks[16 + ln][d0 + 16];
            S0 = __builtin_amdgcn_wmma_f32_16x16x32_bf16(false, qf[c], false, u0.v,
                                                         (short)0, S0, false, false);
            S1 = __builtin_amdgcn_wmma_f32_16x16x32_bf16(false, qf[c], false, u1.v,
                                                         (short)0, S1, false, false);
        }

        // Online softmax (C-layout: VGPR j -> row j + hi*8, lane%16 -> key col)
        float corr[8];
#pragma unroll
        for (int j = 0; j < 8; ++j) {
            const int qg = qrow0 + j + hi * 8;
            float s0 = S0[j], s1 = S1[j];
            if (kb0 + ln > qg)      s0 = -3.0e38f;
            if (kb0 + 16 + ln > qg) s1 = -3.0e38f;
            float mx = fmaxf(s0, s1);
            mx = fmaxf(mx, __shfl_xor(mx, 1, 32));
            mx = fmaxf(mx, __shfl_xor(mx, 2, 32));
            mx = fmaxf(mx, __shfl_xor(mx, 4, 32));
            mx = fmaxf(mx, __shfl_xor(mx, 8, 32));
            const float mnew = fmaxf(mrun[j], mx);
            const float c = __expf(mrun[j] - mnew);
            mrun[j] = mnew;
            const float p0 = __expf(s0 - mnew);
            const float p1 = __expf(s1 - mnew);
            float rs = p0 + p1;
            rs += __shfl_xor(rs, 1, 32);
            rs += __shfl_xor(rs, 2, 32);
            rs += __shfl_xor(rs, 4, 32);
            rs += __shfl_xor(rs, 8, 32);
            lrun[j] = lrun[j] * c + rs;
            corr[j] = c;
            Ps[w][j + hi * 8][ln]      = (__bf16)p0;
            Ps[w][j + hi * 8][16 + ln] = (__bf16)p1;
        }
#pragma unroll
        for (int dt = 0; dt < 8; ++dt)
#pragma unroll
            for (int j = 0; j < 8; ++j)
                acc[dt][j] *= corr[j];

        __syncthreads();

        // O += P @ V  (P as A-fragment from LDS, V^T rows as B-fragments)
        bfu up;
        up.h[0] = *(const v8bf*)&Ps[w][ln][kbA];
        up.h[1] = *(const v8bf*)&Ps[w][ln][kbA + 16];
        const v16bf pf = up.v;
#pragma unroll
        for (int dt = 0; dt < 8; ++dt) {
            bfu uv;
            const int r = dt * 16 + ln;
            uv.h[0] = *(const v8bf*)&Vts[r][kbB];
            uv.h[1] = *(const v8bf*)&Vts[r][kbB + 16];
            acc[dt] = __builtin_amdgcn_wmma_f32_16x16x32_bf16(false, pf, false, uv.v,
                                                              (short)0, acc[dt], false, false);
        }
    }

    // Normalize + store
#pragma unroll
    for (int j = 0; j < 8; ++j) lrun[j] = 1.0f / lrun[j];
#pragma unroll
    for (int dt = 0; dt < 8; ++dt)
#pragma unroll
        for (int j = 0; j < 8; ++j) {
            const int t = qrow0 + j + hi * 8;
            const int d = dt * 16 + ln;
            Yb[((size_t)(b * T_SZ + t)) * C_SZ + h * D_SZ + d] =
                (__bf16)(acc[dt][j] * lrun[j]);
        }
}

// ---------------------------------------------------------------------------
// Workspace layout (bytes)
// ---------------------------------------------------------------------------
static const size_t OFF_XB   = 0;           // x as bf16                (16 MiB)
static const size_t OFF_WTQ  = 16777216;    // Wt q (bf16, [n][k])      ( 8 MiB)
static const size_t OFF_WTK  = 25165824;
static const size_t OFF_WTV  = 33554432;
static const size_t OFF_WTP  = 41943040;
static const size_t OFF_QBUF = 50331648;    // q proj bf16 [m][n]       (16 MiB)
static const size_t OFF_KBUF = 67108864;
static const size_t OFF_VBUF = 83886080;
static const size_t OFF_QH   = 100663296;   // Q roped, [bh][t][d]      (16 MiB)
static const size_t OFF_KH   = 117440512;
static const size_t OFF_VT   = 134217728;   // V^T [bh][d][t]           (16 MiB)
static const size_t OFF_Y    = OFF_QBUF;    // attention out aliases dead qbuf

extern "C" void kernel_launch(void* const* d_in, const int* in_sizes, int n_in,
                              void* d_out, int out_size, void* d_ws, size_t ws_size,
                              hipStream_t stream) {
    (void)in_sizes; (void)n_in; (void)out_size; (void)ws_size;
    const float* x    = (const float*)d_in[0];
    const float* q_w  = (const float*)d_in[1];
    const float* q_b  = (const float*)d_in[2];
    const float* q_A  = (const float*)d_in[3];
    const float* q_Bm = (const float*)d_in[4];
    const float* k_w  = (const float*)d_in[5];
    const float* k_b  = (const float*)d_in[6];
    const float* k_A  = (const float*)d_in[7];
    const float* k_Bm = (const float*)d_in[8];
    const float* v_w  = (const float*)d_in[9];
    const float* v_b  = (const float*)d_in[10];
    const float* v_A  = (const float*)d_in[11];
    const float* v_Bm = (const float*)d_in[12];
    const float* p_w  = (const float*)d_in[13];
    const float* p_b  = (const float*)d_in[14];
    const float* p_A  = (const float*)d_in[15];
    const float* p_Bm = (const float*)d_in[16];
    const float* cosb = (const float*)d_in[17];
    const float* sinb = (const float*)d_in[18];
    const int* use_lora = (const int*)d_in[20];   // d_in[19] = mask (causal, implied)

    char* ws = (char*)d_ws;
    __bf16* Xb   = (__bf16*)(ws + OFF_XB);
    __bf16* Wtq  = (__bf16*)(ws + OFF_WTQ);
    __bf16* Wtk  = (__bf16*)(ws + OFF_WTK);
    __bf16* Wtv  = (__bf16*)(ws + OFF_WTV);
    __bf16* Wtp  = (__bf16*)(ws + OFF_WTP);
    __bf16* Qbuf = (__bf16*)(ws + OFF_QBUF);
    __bf16* Kbuf = (__bf16*)(ws + OFF_KBUF);
    __bf16* Vbuf = (__bf16*)(ws + OFF_VBUF);
    __bf16* Qh   = (__bf16*)(ws + OFF_QH);
    __bf16* Kh   = (__bf16*)(ws + OFF_KH);
    __bf16* Vt   = (__bf16*)(ws + OFF_VT);
    __bf16* Yb   = (__bf16*)(ws + OFF_Y);
    float*  out  = (float*)d_out;

    // 1) x -> bf16
    cvt_f32_bf16<<<M_ROWS * C_SZ / 256, 256, 0, stream>>>(x, Xb);

    // 2) fold LoRA into transposed bf16 weights
    const int prep_blocks = C_SZ * C_SZ / 256;
    prep_w<<<prep_blocks, 256, 0, stream>>>(q_w, q_A, q_Bm, use_lora, Wtq);
    prep_w<<<prep_blocks, 256, 0, stream>>>(k_w, k_A, k_Bm, use_lora, Wtk);
    prep_w<<<prep_blocks, 256, 0, stream>>>(v_w, v_A, v_Bm, use_lora, Wtv);
    prep_w<<<prep_blocks, 256, 0, stream>>>(p_w, p_A, p_Bm, use_lora, Wtp);

    // 3) q/k/v projections (WMMA GEMM, bf16 out)
    dim3 ggrid(C_SZ / 128, M_ROWS / 128);
    gemm_bf16<__bf16><<<ggrid, 256, 0, stream>>>(Xb, Wtq, q_b, Qbuf, M_ROWS, C_SZ, C_SZ);
    gemm_bf16<__bf16><<<ggrid, 256, 0, stream>>>(Xb, Wtk, k_b, Kbuf, M_ROWS, C_SZ, C_SZ);
    gemm_bf16<__bf16><<<ggrid, 256, 0, stream>>>(Xb, Wtv, v_b, Vbuf, M_ROWS, C_SZ, C_SZ);

    // 4) RoPE (1/sqrt(D) folded into Q) + V transpose
    const int rope_blocks = B_SZ * T_SZ * H_SZ * (D_SZ / 2) / 256;
    rope_kernel<<<rope_blocks, 256, 0, stream>>>(Qbuf, cosb, sinb, Qh, 0.08838834764831845f);
    rope_kernel<<<rope_blocks, 256, 0, stream>>>(Kbuf, cosb, sinb, Kh, 1.0f);
    v_transpose<<<B_SZ * H_SZ * D_SZ * T_SZ / 256, 256, 0, stream>>>(Vbuf, Vt);

    // 5) flash attention (WMMA QK^T and AV)
    attn_kernel<<<dim3(T_SZ / 128, B_SZ * H_SZ), 256, 0, stream>>>(Qh, Kh, Vt, Yb);

    // 6) output projection -> fp32 d_out
    gemm_bf16<float><<<ggrid, 256, 0, stream>>>(Yb, Wtp, p_b, out, M_ROWS, C_SZ, C_SZ);
}
